// EpisodeTransformerLayer_68616397521470
// MI455X (gfx1250) — matrix operand order repaired and checked by reading, not compiled, not run on hardware
//
#include <hip/hip_runtime.h>

// ---------------------------------------------------------------------------
// MI455X (gfx1250) transformer layer, bf16 WMMA everywhere.
// Compute-bound (~275 GFLOP vs ~150MB traffic -> ~6.5us @ 23.3TB/s), so all
// six matmuls (QKV, QK^T, PV, Wo, W1, W2) go through v_wmma_f32_16x16x32_bf16
// with fp32 accumulation. Weights (48MB) are L2-resident (192MB global L2).
// Wave32 / WGP: 256-thread blocks = 8 waves.
// CDNA5 data movement: double-buffered LDS tiles; layout-preserving tiles use
// global_load_async_to_lds_b128 (ASYNCcnt) so copies run behind the WMMAs;
// transposing tiles use regs->ds_store overlapped with compute; streaming
// weight reads are warmed with global_prefetch.
// ---------------------------------------------------------------------------

typedef unsigned short u16;
typedef unsigned int   u32;

typedef __attribute__((ext_vector_type(16))) __bf16 v16bf;
typedef __attribute__((ext_vector_type(8)))  float  v8f;

struct __align__(16) B128 { u32 x, y, z, w; };
union Frag16 { B128 u[2]; v16bf v; };

__device__ __forceinline__ u16 f2bf(float f) {
  u32 u = __builtin_bit_cast(u32, f);
  u += 0x7FFFu + ((u >> 16) & 1u);           // round-to-nearest-even
  return (u16)(u >> 16);
}
__device__ __forceinline__ float bf2f(u16 h) {
  u32 u = ((u32)h) << 16;
  return __builtin_bit_cast(float, u);
}
__device__ __forceinline__ v8f wmma_bf16(const v16bf& a, const v16bf& b, v8f c) {
  return __builtin_amdgcn_wmma_f32_16x16x32_bf16(false, a, false, b, (short)0, c,
                                                 false, false);
}

// Async global->LDS copy, 16B per lane (GLOBAL_LOAD_ASYNC_TO_LDS_B128, GV
// mode). lds_off = wave-relative LDS byte address (low 32 bits of the generic
// pointer land in the LDS aperture offset). Tracked by ASYNCcnt.
__device__ __forceinline__ void async_copy_b128(u32 lds_off, const void* gptr) {
  asm volatile("global_load_async_to_lds_b128 %0, %1, off"
               :: "v"(lds_off), "v"(gptr) : "memory");
}
__device__ __forceinline__ void wait_async0() {
#if __has_builtin(__builtin_amdgcn_s_wait_asynccnt)
  __builtin_amdgcn_s_wait_asynccnt(0);
#else
  asm volatile("s_wait_asynccnt 0x0" ::: "memory");
#endif
}

#define DIMD 1024
#define NH   16
#define HDIM 64
#define FFD  4096
#define BB   4
#define TT   2048
#define ROWS (BB * TT)   // 8192

// ---------------------------------------------------------------------------
// fp32 -> bf16 elementwise conversion (weights, done once per launch)
// ---------------------------------------------------------------------------
__global__ __launch_bounds__(256)
void k_f32_to_bf16(const float* __restrict__ in, u16* __restrict__ out, int n) {
  int i = blockIdx.x * 256 + threadIdx.x;
  if (i < n) out[i] = f2bf(in[i]);
}

// ---------------------------------------------------------------------------
// LayerNorm over rows of 1024, output bf16. One block per row.
// ---------------------------------------------------------------------------
__global__ __launch_bounds__(256)
void k_layernorm_bf16(const float* __restrict__ x, const float* __restrict__ g,
                      const float* __restrict__ b, u16* __restrict__ out) {
  __shared__ float red[256];
  const int row = blockIdx.x, tid = threadIdx.x;
  const float* xr = x + (size_t)row * DIMD;
  float v[4], s = 0.f, s2 = 0.f;
#pragma unroll
  for (int i = 0; i < 4; ++i) {
    v[i] = xr[tid + i * 256];
    s += v[i]; s2 += v[i] * v[i];
  }
  red[tid] = s; __syncthreads();
  for (int o = 128; o > 0; o >>= 1) { if (tid < o) red[tid] += red[tid + o]; __syncthreads(); }
  const float mean = red[0] * (1.f / DIMD);
  __syncthreads();
  red[tid] = s2; __syncthreads();
  for (int o = 128; o > 0; o >>= 1) { if (tid < o) red[tid] += red[tid + o]; __syncthreads(); }
  const float var  = red[0] * (1.f / DIMD) - mean * mean;
  const float rstd = rsqrtf(var + 1e-5f);
  u16* orow = out + (size_t)row * DIMD;
#pragma unroll
  for (int i = 0; i < 4; ++i) {
    const int c = tid + i * 256;
    orow[c] = f2bf((v[i] - mean) * rstd * g[c] + b[c]);
  }
}

// ---------------------------------------------------------------------------
// Tiled bf16 GEMM: C(M,N) = A(M,K) @ B(K,N) + bias, epilogue variants.
// Block tile 64x128, K-tile 32; 8 waves as 2x4, each wave a 32x32 tile
// (4 x v_wmma_f32_16x16x32_bf16 per K step). Double-buffered LDS:
//  - A tile (layout-preserving) via global_load_async_to_lds_b128
//  - B tile via regs -> transposed ds_store, committed after the WMMAs
// One barrier per K step; B stream prefetched 2 tiles ahead into GL2.
// EPI: 0 = store bf16, 1 = exact GeLU -> bf16, 2 = residual add -> fp32
// ---------------------------------------------------------------------------
template <int EPI>
__global__ __launch_bounds__(256)
void k_gemm_bf16(const u16* __restrict__ A, const u16* __restrict__ B,
                 const float* __restrict__ bias, const float* __restrict__ resid,
                 void* __restrict__ outp, int M, int N, int K) {
  constexpr int BM = 64, BN = 128, BK = 32, LDA = BK + 8, LDB = BK + 8;
  __shared__ u16 As[2][BM * LDA];     // row-major [m][k], padded
  __shared__ u16 Bs[2][BN * LDB];     // transposed [n][k], padded
  const int tid = threadIdx.x, wave = tid >> 5, lane = tid & 31;
  const int l16 = lane & 15, lhalf = lane >> 4;
  const int m0 = blockIdx.y * BM, n0 = blockIdx.x * BN;
  const int wm = (wave >> 2) * 32, wn = (wave & 3) * 32;
  const v8f vz = {};
  v8f acc[2][2] = {{vz, vz}, {vz, vz}};

  // per-thread tile-copy coordinates
  const int am = (tid * 8) >> 5, ak = (tid * 8) & 31;            // A: 1 x b128
  const int bk = (tid * 8) >> 7, bn = (tid * 8) & 127;           // B: 2 x b128 (k, k+16)
  const u16* aptr = A + (size_t)(m0 + am) * K + ak;
  const u32 alds[2] = {(u32)(size_t)&As[0][am * LDA + ak],
                       (u32)(size_t)&As[1][am * LDA + ak]};

  // prologue: stage tile 0
  async_copy_b128(alds[0], aptr);
  {
    B128 t0 = *(const B128*)(B + (size_t)bk * N + n0 + bn);
    B128 t1 = *(const B128*)(B + (size_t)(bk + 16) * N + n0 + bn);
    const u16* p0 = (const u16*)&t0; const u16* p1 = (const u16*)&t1;
#pragma unroll
    for (int j = 0; j < 8; ++j) {
      Bs[0][(bn + j) * LDB + bk] = p0[j];
      Bs[0][(bn + j) * LDB + bk + 16] = p1[j];
    }
  }
  wait_async0();
  __syncthreads();

  const int nK = K / BK;
  for (int kt = 0; kt < nK; ++kt) {
    const int cur = kt & 1;
    const bool more = (kt + 1 < nK);
    B128 t0, t1;
    if (more) {  // issue next-tile copies before computing on current tile
      async_copy_b128(alds[1 - cur], aptr + (size_t)(kt + 1) * BK);
      if (kt + 2 < nK)
        __builtin_prefetch(B + (size_t)((kt + 2) * BK + bk) * N + n0 + bn, 0, 3);
      t0 = *(const B128*)(B + (size_t)((kt + 1) * BK + bk) * N + n0 + bn);
      t1 = *(const B128*)(B + (size_t)((kt + 1) * BK + bk + 16) * N + n0 + bn);
    }

    Frag16 af[2], bfr[2];
#pragma unroll
    for (int i = 0; i < 2; ++i) {  // A frag: lane m, K 0-15 / 16-31 contiguous
      const u16* pa = &As[cur][(wm + i * 16 + l16) * LDA + lhalf * 16];
      af[i].u[0] = *(const B128*)(pa);
      af[i].u[1] = *(const B128*)(pa + 8);
    }
#pragma unroll
    for (int j = 0; j < 2; ++j) {  // B frag: lane n, K 0-15 / 16-31 contiguous
      const u16* pb = &Bs[cur][(wn + j * 16 + l16) * LDB + lhalf * 16];
      bfr[j].u[0] = *(const B128*)(pb);
      bfr[j].u[1] = *(const B128*)(pb + 8);
    }
#pragma unroll
    for (int i = 0; i < 2; ++i)
#pragma unroll
      for (int j = 0; j < 2; ++j)
        acc[i][j] = wmma_bf16(af[i].v, bfr[j].v, acc[i][j]);

    if (more) {  // commit next B tile (transposed) after the WMMAs
      const u16* p0 = (const u16*)&t0; const u16* p1 = (const u16*)&t1;
#pragma unroll
      for (int j = 0; j < 8; ++j) {
        Bs[1 - cur][(bn + j) * LDB + bk] = p0[j];
        Bs[1 - cur][(bn + j) * LDB + bk + 16] = p1[j];
      }
      wait_async0();
    }
    __syncthreads();
  }

#pragma unroll
  for (int j = 0; j < 2; ++j) {
    const int col = n0 + wn + j * 16 + l16;
    const float bv = bias[col];
#pragma unroll
    for (int i = 0; i < 2; ++i) {
#pragma unroll
      for (int r = 0; r < 8; ++r) {
        const int row = m0 + wm + i * 16 + lhalf * 8 + r;   // C layout
        const float val = acc[i][j][r] + bv;
        const size_t off = (size_t)row * N + col;
        if (EPI == 0) {
          ((u16*)outp)[off] = f2bf(val);
        } else if (EPI == 1) {
          const float gel = 0.5f * val * (1.f + erff(val * 0.70710678118654752f));
          ((u16*)outp)[off] = f2bf(gel);
        } else {
          ((float*)outp)[off] = resid[off] + val;
        }
      }
    }
  }
}

// ---------------------------------------------------------------------------
// RoPE + head split: qkv(B,T,3,H,HD) bf16 -> q/k/v in (B,H,T,HD) bf16.
// ---------------------------------------------------------------------------
__global__ __launch_bounds__(256)
void k_rope_split(const u16* __restrict__ qkv, const float* __restrict__ cosb,
                  const float* __restrict__ sinb, u16* __restrict__ qh,
                  u16* __restrict__ kh, u16* __restrict__ vh) {
  const int idx = blockIdx.x * 256 + threadIdx.x;   // BB*NH*TT*32
  const int i = idx & 31;
  const int t = (idx >> 5) & (TT - 1);
  const int h = (idx >> 16) & (NH - 1);
  const int b = idx >> 20;
  const size_t src = (size_t)(b * TT + t) * (3 * DIMD) + h * HDIM;
  const float c = cosb[t * 32 + i], s = sinb[t * 32 + i];
  const float q1 = bf2f(qkv[src + i]),        q2 = bf2f(qkv[src + 32 + i]);
  const float k1 = bf2f(qkv[src + 1024 + i]), k2 = bf2f(qkv[src + 1056 + i]);
  const size_t dst = (size_t)((b * NH + h) * TT + t) * HDIM;
  qh[dst + i]      = f2bf(q1 * c - q2 * s);
  qh[dst + 32 + i] = f2bf(q2 * c + q1 * s);
  kh[dst + i]      = f2bf(k1 * c - k2 * s);
  kh[dst + 32 + i] = f2bf(k2 * c + k1 * s);
  vh[dst + i]      = qkv[src + 2048 + i];
  vh[dst + 32 + i] = qkv[src + 2080 + i];
}

// ---------------------------------------------------------------------------
// Flash attention: block = (b, h, 128-query tile); 8 waves x 16 queries.
// Double-buffered 32-key K/V tiles: K (layout-preserving) streamed with
// global_load_async_to_lds_b128; V loaded to regs and committed transposed
// after the WMMAs. QK^T: 4 WMMAs; online softmax (shfl_xor row reductions
// inside the 16-lane half matching the C layout); P re-striped C->A via
// per-wave LDS; PV: 4 WMMAs.
// ---------------------------------------------------------------------------
__global__ __launch_bounds__(256)
void k_attention(const u16* __restrict__ Q, const u16* __restrict__ Kmat,
                 const u16* __restrict__ Vmat, u16* __restrict__ Oc) {
  constexpr int LDK = HDIM + 8;  // 72 halfwords: bank-spread, keeps 16B align
  constexpr int LDV = 32 + 8;    // 40
  constexpr int LDP = 32 + 8;    // 40
  __shared__ u16 Ks[2][32 * LDK];       // [key][hd]
  __shared__ u16 Vs[2][HDIM * LDV];     // transposed: [hd][key]
  __shared__ u16 Ps[8][16 * LDP];       // per-wave P tile 16x32
  const int tid = threadIdx.x, wave = tid >> 5, lane = tid & 31;
  const int l16 = lane & 15, lhalf = lane >> 4;
  const int qb = blockIdx.x & 15, bh = blockIdx.x >> 4;   // T/128=16 tiles
  const size_t base = (size_t)bh * TT * HDIM;
  const int q0 = qb * 128 + wave * 16;
  const float scale = 0.125f;           // 1/sqrt(64)

  Frag16 qf[2];                         // Q rows q0..q0+15, K-dim = hd (0..63)
  {
    const u16* qp = Q + base + (size_t)(q0 + l16) * HDIM + lhalf * 16;
    qf[0].u[0] = *(const B128*)(qp);
    qf[0].u[1] = *(const B128*)(qp + 8);
    qf[1].u[0] = *(const B128*)(qp + 32);
    qf[1].u[1] = *(const B128*)(qp + 40);
  }
  const v8f vz = {};
  v8f o0 = vz, o1 = vz, o2 = vz, o3 = vz;   // O tile 16x64 fp32
  float m[8], l[8];
#pragma unroll
  for (int r = 0; r < 8; ++r) { m[r] = -3.0e38f; l[r] = 0.f; }

  // per-thread tile-copy coordinates (32x64 bf16 tile, 1 x b128 each)
  const int kr = (tid * 8) >> 6, kc = (tid * 8) & 63;
  const u32 klds[2] = {(u32)(size_t)&Ks[0][kr * LDK + kc],
                       (u32)(size_t)&Ks[1][kr * LDK + kc]};

  // prologue: stage key-tile 0
  async_copy_b128(klds[0], Kmat + base + (size_t)kr * HDIM + kc);
  {
    B128 tv = *(const B128*)(Vmat + base + (size_t)kr * HDIM + kc);
    const u16* pv = (const u16*)&tv;
#pragma unroll
    for (int j = 0; j < 8; ++j) Vs[0][(kc + j) * LDV + kr] = pv[j];
  }
  wait_async0();
  __syncthreads();

  for (int kt = 0; kt < TT / 32; ++kt) {
    const int cur = kt & 1;
    const bool more = (kt + 1 < TT / 32);
    B128 tv;
    if (more) {
      const size_t gnext = base + (size_t)((kt + 1) * 32 + kr) * HDIM + kc;
      async_copy_b128(klds[1 - cur], Kmat + gnext);
      if (kt + 2 < TT / 32) {
        __builtin_prefetch(Kmat + base + (size_t)((kt + 2) * 32 + kr) * HDIM + kc, 0, 3);
        __builtin_prefetch(Vmat + base + (size_t)((kt + 2) * 32 + kr) * HDIM + kc, 0, 3);
      }
      tv = *(const B128*)(Vmat + gnext);
    }

    // S = scale * Q @ K^T  (B frag: lane n = key, K-dim = hd contiguous)
    v8f s0 = vz, s1 = vz;
    {
      Frag16 kf;
      const u16* pk0 = &Ks[cur][l16 * LDK + lhalf * 16];
      kf.u[0] = *(const B128*)(pk0);
      kf.u[1] = *(const B128*)(pk0 + 8);
      s0 = wmma_bf16(qf[0].v, kf.v, s0);
      kf.u[0] = *(const B128*)(pk0 + 32);
      kf.u[1] = *(const B128*)(pk0 + 40);
      s0 = wmma_bf16(qf[1].v, kf.v, s0);
      const u16* pk1 = &Ks[cur][(16 + l16) * LDK + lhalf * 16];
      kf.u[0] = *(const B128*)(pk1);
      kf.u[1] = *(const B128*)(pk1 + 8);
      s1 = wmma_bf16(qf[0].v, kf.v, s1);
      kf.u[0] = *(const B128*)(pk1 + 32);
      kf.u[1] = *(const B128*)(pk1 + 40);
      s1 = wmma_bf16(qf[1].v, kf.v, s1);
    }

    // online softmax; write P (bf16) into per-wave LDS, C-layout -> row-major
#pragma unroll
    for (int r = 0; r < 8; ++r) {
      const float a0 = s0[r] * scale, a1 = s1[r] * scale;
      float amax = fmaxf(a0, a1);
#pragma unroll
      for (int off = 1; off < 16; off <<= 1) amax = fmaxf(amax, __shfl_xor(amax, off, 32));
      const float mn = fmaxf(m[r], amax);
      const float corr = __expf(m[r] - mn);
      m[r] = mn;
      const float e0 = __expf(a0 - mn), e1 = __expf(a1 - mn);
      float rs = e0 + e1;
#pragma unroll
      for (int off = 1; off < 16; off <<= 1) rs += __shfl_xor(rs, off, 32);
      l[r] = l[r] * corr + rs;
      o0[r] *= corr; o1[r] *= corr; o2[r] *= corr; o3[r] *= corr;
      Ps[wave][(lhalf * 8 + r) * LDP + l16] = f2bf(e0);
      Ps[wave][(lhalf * 8 + r) * LDP + 16 + l16] = f2bf(e1);
    }
    asm volatile("" ::: "memory");   // keep DS reads after DS writes (in-order HW)

    // O += P @ V  (A frag from Ps; B frag from transposed Vs: keys contiguous)
    Frag16 pf;
    {
      const u16* pp = &Ps[wave][l16 * LDP + lhalf * 16];
      pf.u[0] = *(const B128*)(pp);
      pf.u[1] = *(const B128*)(pp + 8);
    }
    {
      Frag16 vf;
      const u16* pv0 = &Vs[cur][l16 * LDV + lhalf * 16];
      vf.u[0] = *(const B128*)(pv0); vf.u[1] = *(const B128*)(pv0 + 8);
      o0 = wmma_bf16(pf.v, vf.v, o0);
      const u16* pv1 = &Vs[cur][(16 + l16) * LDV + lhalf * 16];
      vf.u[0] = *(const B128*)(pv1); vf.u[1] = *(const B128*)(pv1 + 8);
      o1 = wmma_bf16(pf.v, vf.v, o1);
      const u16* pv2 = &Vs[cur][(32 + l16) * LDV + lhalf * 16];
      vf.u[0] = *(const B128*)(pv2); vf.u[1] = *(const B128*)(pv2 + 8);
      o2 = wmma_bf16(pf.v, vf.v, o2);
      const u16* pv3 = &Vs[cur][(48 + l16) * LDV + lhalf * 16];
      vf.u[0] = *(const B128*)(pv3); vf.u[1] = *(const B128*)(pv3 + 8);
      o3 = wmma_bf16(pf.v, vf.v, o3);
    }

    if (more) {  // commit next V tile (transposed) after the WMMAs
      const u16* pv = (const u16*)&tv;
#pragma unroll
      for (int j = 0; j < 8; ++j) Vs[1 - cur][(kc + j) * LDV + kr] = pv[j];
      wait_async0();
    }
    __syncthreads();
  }

  // normalize and scatter back to (B,T,D) bf16
  const int b = bh >> 4, h = bh & 15;
#pragma unroll
  for (int r = 0; r < 8; ++r) {
    const float inv = 1.f / l[r];
    const int t = q0 + lhalf * 8 + r;
    u16* orow = Oc + (size_t)(b * TT + t) * DIMD + h * HDIM;
    orow[l16]      = f2bf(o0[r] * inv);
    orow[16 + l16] = f2bf(o1[r] * inv);
    orow[32 + l16] = f2bf(o2[r] * inv);
    orow[48 + l16] = f2bf(o3[r] * inv);
  }
}

// ---------------------------------------------------------------------------
// Workspace layout (bytes). Dead regions are reused; peak ~152 MiB.
// ---------------------------------------------------------------------------
static const size_t OFF_WQKV = 0;                                        // bf16 1024x3072
static const size_t OFF_WO   = OFF_WQKV + (size_t)DIMD * 3 * DIMD * 2;   // bf16 1024x1024
static const size_t OFF_W1   = OFF_WO   + (size_t)DIMD * DIMD * 2;       // bf16 1024x4096
static const size_t OFF_W2   = OFF_W1   + (size_t)DIMD * FFD * 2;        // bf16 4096x1024
static const size_t OFF_H1   = OFF_W2   + (size_t)FFD * DIMD * 2;        // bf16 8192x1024
static const size_t OFF_QKV  = OFF_H1   + (size_t)ROWS * DIMD * 2;       // bf16 8192x3072
static const size_t OFF_QH   = OFF_QKV  + (size_t)ROWS * 3 * DIMD * 2;   // bf16 (B,H,T,HD)
static const size_t OFF_KH   = OFF_QH   + (size_t)ROWS * DIMD * 2;
static const size_t OFF_VH   = OFF_KH   + (size_t)ROWS * DIMD * 2;
// reuse of dead regions:
static const size_t OFF_ATTN = OFF_H1;                                   // bf16 8192x1024
static const size_t OFF_X1   = OFF_QKV;                                  // f32  8192x1024
static const size_t OFF_H2   = OFF_QKV + (size_t)ROWS * DIMD * 4;        // bf16 8192x1024
static const size_t OFF_FF   = OFF_QH;                                   // bf16 8192x4096

extern "C" void kernel_launch(void* const* d_in, const int* in_sizes, int n_in,
                              void* d_out, int out_size, void* d_ws, size_t ws_size,
                              hipStream_t stream) {
  (void)in_sizes; (void)n_in; (void)out_size; (void)ws_size;
  const float* x        = (const float*)d_in[0];
  const float* rope_cos = (const float*)d_in[1];
  const float* rope_sin = (const float*)d_in[2];
  const float* ln1_g    = (const float*)d_in[3];
  const float* ln1_b    = (const float*)d_in[4];
  const float* Wqkv     = (const float*)d_in[5];
  const float* bqkv     = (const float*)d_in[6];
  const float* Wo       = (const float*)d_in[7];
  const float* bo       = (const float*)d_in[8];
  const float* ln2_g    = (const float*)d_in[9];
  const float* ln2_b    = (const float*)d_in[10];
  const float* W1       = (const float*)d_in[11];
  const float* b1       = (const float*)d_in[12];
  const float* W2       = (const float*)d_in[13];
  const float* b2       = (const float*)d_in[14];

  char* ws = (char*)d_ws;
  u16*   wqkv_bf = (u16*)(ws + OFF_WQKV);
  u16*   wo_bf   = (u16*)(ws + OFF_WO);
  u16*   w1_bf   = (u16*)(ws + OFF_W1);
  u16*   w2_bf   = (u16*)(ws + OFF_W2);
  u16*   h1      = (u16*)(ws + OFF_H1);
  u16*   qkv     = (u16*)(ws + OFF_QKV);
  u16*   qh      = (u16*)(ws + OFF_QH);
  u16*   kh      = (u16*)(ws + OFF_KH);
  u16*   vh      = (u16*)(ws + OFF_VH);
  u16*   attn_c  = (u16*)(ws + OFF_ATTN);
  float* x1      = (float*)(ws + OFF_X1);
  u16*   h2      = (u16*)(ws + OFF_H2);
  u16*   ff      = (u16*)(ws + OFF_FF);

  // 1) weights fp32 -> bf16 (L2-resident thereafter)
  {
    const int nq = DIMD * 3 * DIMD, no = DIMD * DIMD, n1 = DIMD * FFD, n2 = FFD * DIMD;
    k_f32_to_bf16<<<(nq + 255) / 256, 256, 0, stream>>>(Wqkv, wqkv_bf, nq);
    k_f32_to_bf16<<<(no + 255) / 256, 256, 0, stream>>>(Wo,   wo_bf,   no);
    k_f32_to_bf16<<<(n1 + 255) / 256, 256, 0, stream>>>(W1,   w1_bf,   n1);
    k_f32_to_bf16<<<(n2 + 255) / 256, 256, 0, stream>>>(W2,   w2_bf,   n2);
  }

  // 2) h1 = LN1(x)
  k_layernorm_bf16<<<ROWS, 256, 0, stream>>>(x, ln1_g, ln1_b, h1);

  // 3) qkv = h1 @ Wqkv + bqkv   (8192x1024 @ 1024x3072)
  k_gemm_bf16<0><<<dim3(3 * DIMD / 128, ROWS / 64), 256, 0, stream>>>(
      h1, wqkv_bf, bqkv, nullptr, qkv, ROWS, 3 * DIMD, DIMD);

  // 4) RoPE on q,k + split into (B,H,T,HD)
  k_rope_split<<<(BB * NH * TT * 32) / 256, 256, 0, stream>>>(
      qkv, rope_cos, rope_sin, qh, kh, vh);

  // 5) flash attention -> attn_c (B,T,D) bf16
  k_attention<<<BB * NH * (TT / 128), 256, 0, stream>>>(qh, kh, vh, attn_c);

  // 6) x1 = x + attn_c @ Wo + bo  (fp32 residual stream)
  k_gemm_bf16<2><<<dim3(DIMD / 128, ROWS / 64), 256, 0, stream>>>(
      attn_c, wo_bf, bo, x, x1, ROWS, DIMD, DIMD);

  // 7) h2 = LN2(x1)
  k_layernorm_bf16<<<ROWS, 256, 0, stream>>>(x1, ln2_g, ln2_b, h2);

  // 8) ff = gelu(h2 @ W1 + b1)   (8192x1024 @ 1024x4096)
  k_gemm_bf16<1><<<dim3(FFD / 128, ROWS / 64), 256, 0, stream>>>(
      h2, w1_bf, b1, nullptr, ff, ROWS, FFD, DIMD);

  // 9) out = x1 + ff @ W2 + b2   (8192x4096 @ 4096x1024)
  k_gemm_bf16<2><<<dim3(DIMD / 128, ROWS / 64), 256, 0, stream>>>(
      ff, w2_bf, b2, x1, d_out, ROWS, DIMD, FFD);
}
